// MultiheadAttention_75677323755700
// MI455X (gfx1250) — matrix operand order repaired and checked by reading, not compiled, not run on hardware
//
#include <hip/hip_runtime.h>
#include <hip/hip_bf16.h>
#include <math.h>

// ---------------------------------------------------------------------------
// MultiheadAttention (B=2, S=1024, D=1024, 64 chunks of head-dim 16)
// GEMMs: v_wmma_f32_16x16x32_bf16 fed by TDM tensor_load_to_lds (double
// buffered); softmax in fp32. Outputs: out [B,S,D] fp32 + weights [B,C,S,S]
// fp32 concatenated in d_out.
// ---------------------------------------------------------------------------

typedef __attribute__((ext_vector_type(16))) __bf16 v16bf;
typedef __attribute__((ext_vector_type(8)))  __bf16 v8bf;
typedef __attribute__((ext_vector_type(8)))  float  v8f;
typedef __attribute__((ext_vector_type(4)))  unsigned int v4u;
typedef __attribute__((ext_vector_type(8)))  int    v8i;
typedef __attribute__((ext_vector_type(4)))  int    v4i;

#define NB 2       // batch
#define SS 1024    // sequence
#define DD 1024    // model dim
#define CC 64      // chunks ("heads" of the quirky split)
#define HH 16      // per-chunk dim
#define HP 32      // padded per-chunk dim (zeros in [16,32)) for K=32 WMMA
#define MR (NB*SS) // 2048 rows in the flattened [B*S, D] GEMMs

#if defined(__gfx1250__) && __has_builtin(__builtin_amdgcn_tensor_load_to_lds) \
    && __has_builtin(__builtin_amdgcn_s_wait_tensorcnt)
#define HAVE_TDM 1
#else
#define HAVE_TDM 0
#endif

static __device__ __forceinline__ v8f wmma_bf16(v16bf a, v16bf b, v8f c) {
  // D = A(16x32) * B(32x16) + C, fp32 accumulate
  return __builtin_amdgcn_wmma_f32_16x16x32_bf16(
      /*neg_a=*/false, a, /*neg_b=*/false, b,
      /*c_mod=*/(short)0, c, /*reuse_a=*/false, /*reuse_b=*/false);
}

#if HAVE_TDM
// Issue a 2-D TDM load: tile (tile1 rows x tile0 elems) of a bf16 tensor with
// row stride `stride0` starting at `gptr`, into LDS at byte offset `lds_off`.
// D# layout per CDNA5 ISA ch.7.3/8 (group0 128b, group1 256b, groups 2/3
// zero => 2-D tensor). This toolchain's builtin takes 6 args:
// (uint32x4 g0, int32x8 g1, int32x4 g2, int32x4 g3, int32x8 extra, i32 cpol).
static __device__ __forceinline__ void tdm_load_2d_bf16(
    const __bf16* gptr, unsigned lds_off, unsigned dim0, unsigned dim1,
    unsigned stride0, unsigned tile0, unsigned tile1) {
  unsigned long long ga = (unsigned long long)gptr;
  v4u g0;
  g0.x = 1u;                                   // count=1, user descriptor
  g0.y = lds_off;                              // lds_addr [63:32]
  g0.z = (unsigned)(ga & 0xffffffffu);         // global_addr[31:0]
  g0.w = (unsigned)((ga >> 32) & 0x01ffffffu)  // global_addr[56:32]
         | (2u << 30);                         // type=2 ("image")
  v8i g1;
  g1[0] = (int)(1u << 16);                     // data_size=1 (2 bytes/elem)
  g1[1] = (int)((dim0 & 0xffffu) << 16);       // tensor_dim0[15:0] @ [63:48]
  g1[2] = (int)((dim0 >> 16) | ((dim1 & 0xffffu) << 16)); // dim0 hi, dim1 lo
  g1[3] = (int)((dim1 >> 16) | (tile0 << 16)); // dim1 hi, tile_dim0 @[127:112]
  g1[4] = (int)tile1;                          // tile_dim1; tile_dim2=0
  g1[5] = (int)stride0;                        // tensor_dim0_stride[31:0]
  g1[6] = 0;                                   // stride0[47:32]=0, stride1 lo=0
  g1[7] = 0;                                   // stride1 hi=0
  v4i z4 = {0, 0, 0, 0};
  v8i z8 = {0, 0, 0, 0, 0, 0, 0, 0};
  __builtin_amdgcn_tensor_load_to_lds(g0, g1, z4, z4, z8, 0);
}
static __device__ __forceinline__ unsigned lds_offset_of(const void* p) {
  // low 32 bits of a generic pointer into LDS == byte offset in group segment
  return (unsigned)(unsigned long long)(uintptr_t)p;
}
#endif

// ---------------------------------------------------------------------------
// Stage 1: fp32 -> bf16 conversion (grid-stride)
// ---------------------------------------------------------------------------
__global__ void cvt_f32_to_bf16(const float* __restrict__ x,
                                __bf16* __restrict__ y, int n) {
  int i = blockIdx.x * blockDim.x + threadIdx.x;
  int stride = gridDim.x * blockDim.x;
  for (; i < n; i += stride) y[i] = (__bf16)x[i];
}

// ---------------------------------------------------------------------------
// Stage 2/5: tiled GEMM  Y = X(bf16, [M x 1024] row-major) @ W^T
//            W is [1024 x 1024] bf16 row-major ([n][k]) -> k contiguous.
// A/B tiles are DMA'd into LDS by the Tensor Data Mover (wave 0 issues the
// descriptors; TENSORcnt + workgroup barrier synchronize), double buffered so
// the next K-step's DMA overlaps this step's 4 WMMAs per wave.
// Epilogue modes:
//   0: fp32 row-major [M x 1024]                (final output GEMM)
//   1: bf16 [B][C][S][HP], h padded with zeros  (Q / K projections)
//   2: bf16 [B][C][HH][S]  (transposed)         (V projection)
// ---------------------------------------------------------------------------
template <int MODE>
__global__ __launch_bounds__(256) void gemm_xt(const __bf16* __restrict__ X,
                                               const __bf16* __restrict__ W,
                                               float* __restrict__ Yf,
                                               __bf16* __restrict__ Yb) {
  __shared__ __bf16 As[2][64][32];
  __shared__ __bf16 Bs[2][128][32];

  const int tid  = threadIdx.x;
  const int wave = tid >> 5;
  const int lane = tid & 31;
  const int half = lane >> 4;      // 0 or 1
  const int nl   = lane & 15;      // N position / M row within tile

  const int m0 = blockIdx.x * 64;
  const int n0 = blockIdx.y * 128;

  const int wm = wave & 3;         // wave row (4)
  const int wn = wave >> 2;        // wave col (2)

  v8f acc[4] = {};

#if HAVE_TDM
  if (wave == 0) {  // prologue: DMA first K-step tiles into buffer 0
    tdm_load_2d_bf16(X + (size_t)m0 * DD, lds_offset_of(&As[0][0][0]),
                     DD, MR - m0, DD, 32, 64);
    tdm_load_2d_bf16(W + (size_t)n0 * DD, lds_offset_of(&Bs[0][0][0]),
                     DD, DD - n0, DD, 32, 128);
  }
  int buf = 0;
  for (int kk = 0; kk < DD; kk += 32) {
    if (wave == 0) __builtin_amdgcn_s_wait_tensorcnt(0);  // buffer `buf` ready
    __syncthreads();
    if (wave == 0 && kk + 32 < DD) {  // prefetch next K-step into other buffer
      tdm_load_2d_bf16(X + (size_t)m0 * DD + kk + 32,
                       lds_offset_of(&As[buf ^ 1][0][0]),
                       DD - kk - 32, MR - m0, DD, 32, 64);
      tdm_load_2d_bf16(W + (size_t)n0 * DD + kk + 32,
                       lds_offset_of(&Bs[buf ^ 1][0][0]),
                       DD - kk - 32, DD - n0, DD, 32, 128);
    }
    // A fragment: row = M (lane 0..15); elems 0-15: K=half*16 .. half*16+15
    v16bf afrag;
    {
      const int row = wm * 16 + nl;
      const int kb  = half * 16;
      ((v8bf*)&afrag)[0] = *(const v8bf*)&As[buf][row][kb];
      ((v8bf*)&afrag)[1] = *(const v8bf*)&As[buf][row][kb + 8];
    }
#pragma unroll
    for (int tn = 0; tn < 4; ++tn) {
      const int ncol = wn * 64 + tn * 16 + nl;
      v16bf bfrag = *(const v16bf*)&Bs[buf][ncol][half * 16];
      acc[tn] = wmma_bf16(afrag, bfrag, acc[tn]);
    }
    buf ^= 1;
  }
#else
  // Fallback: manual staging (single buffered)
  const int ar = tid >> 2, ak = (tid & 3) * 8;
  const int br = tid >> 2, bk = (tid & 3) * 8;
  for (int kk = 0; kk < DD; kk += 32) {
    __syncthreads();
    *(v8bf*)&As[0][ar][ak] = *(const v8bf*)(X + (size_t)(m0 + ar) * DD + kk + ak);
    *(v8bf*)&Bs[0][br][bk] = *(const v8bf*)(W + (size_t)(n0 + br) * DD + kk + bk);
    *(v8bf*)&Bs[0][br + 64][bk] =
        *(const v8bf*)(W + (size_t)(n0 + br + 64) * DD + kk + bk);
    __syncthreads();
    v16bf afrag;
    {
      const int row = wm * 16 + nl;
      const int kb  = half * 16;
      ((v8bf*)&afrag)[0] = *(const v8bf*)&As[0][row][kb];
      ((v8bf*)&afrag)[1] = *(const v8bf*)&As[0][row][kb + 8];
    }
#pragma unroll
    for (int tn = 0; tn < 4; ++tn) {
      const int ncol = wn * 64 + tn * 16 + nl;
      v16bf bfrag = *(const v16bf*)&Bs[0][ncol][half * 16];
      acc[tn] = wmma_bf16(afrag, bfrag, acc[tn]);
    }
  }
#endif

  // Epilogue. C layout: elem r -> M = r + half*8, N = nl.
#pragma unroll
  for (int tn = 0; tn < 4; ++tn) {
#pragma unroll
    for (int r = 0; r < 8; ++r) {
      const float vv = acc[tn][r];
      const int m = m0 + wm * 16 + half * 8 + r;
      const int n = n0 + wn * 64 + tn * 16 + nl;
      if (MODE == 0) {
        Yf[(size_t)m * DD + n] = vv;
      } else if (MODE == 1) {
        const int b = m >> 10, s = m & 1023, ch = n >> 4, h = n & 15;
        const size_t base = (((size_t)b * CC + ch) * SS + s) * HP + h;
        Yb[base]      = (__bf16)vv;
        Yb[base + 16] = (__bf16)0.0f;  // zero the padded half of HP
      } else {  // MODE == 2: [B][C][H][S]
        const int b = m >> 10, s = m & 1023, ch = n >> 4, h = n & 15;
        Yb[(((size_t)b * CC + ch) * HH + h) * SS + s] = (__bf16)vv;
      }
    }
  }
}

// ---------------------------------------------------------------------------
// Stage 3+4 fused: scores -> softmax -> weights(fp32 to d_out) -> attended
// One block per (b, c, 16-query block). 8 waves; wave w owns keys
// [w*128, w*128+128). Row reductions: shfl_xor within 16-lane halves,
// then LDS across the 8 waves.
// ---------------------------------------------------------------------------
__global__ __launch_bounds__(256) void attn_fused(
    const __bf16* __restrict__ Qp,   // [B][C][S][HP] (padded)
    const __bf16* __restrict__ Kp,   // [B][C][S][HP] (padded)
    const __bf16* __restrict__ Vt,   // [B][C][HH][S]
    float* __restrict__ Wout,        // [B][C][S][S]
    __bf16* __restrict__ Merged) {   // [B*S][D] bf16, col = c*16+h
  __shared__ float  red[8][16];
  __shared__ __bf16 wlds[8][16][128];
  __shared__ float  attacc[8][16][16];

  const int q0 = blockIdx.x * 16;
  const int c  = blockIdx.y;
  const int b  = blockIdx.z;

  const int tid  = threadIdx.x;
  const int wave = tid >> 5;
  const int lane = tid & 31;
  const int half = lane >> 4;
  const int nl   = lane & 15;

  const size_t bcBase = ((size_t)b * CC + c) * SS;

  // ---- scores: A = q rows (K = padded h), B = k rows transposed ----
  v16bf aq;
  {
    const __bf16* qrow = Qp + (bcBase + q0 + nl) * HP;
    ((v8bf*)&aq)[0] = *(const v8bf*)(qrow + half * 16);
    ((v8bf*)&aq)[1] = *(const v8bf*)(qrow + half * 16 + 8);
  }

  v8f acc[8] = {};
#pragma unroll
  for (int t = 0; t < 8; ++t) {
    const int kkey = wave * 128 + t * 16 + nl;            // key index = N
    const __bf16* krow = Kp + (bcBase + kkey) * HP + half * 16;
    v16bf bf = *(const v16bf*)krow;                       // K contiguous
    acc[t] = wmma_bf16(aq, bf, acc[t]);
  }

  // scale by 1/sqrt(16)
#pragma unroll
  for (int t = 0; t < 8; ++t)
#pragma unroll
    for (int r = 0; r < 8; ++r) acc[t][r] *= 0.25f;

  // ---- row max ----
  float rm[8];
#pragma unroll
  for (int r = 0; r < 8; ++r) {
    float m = acc[0][r];
#pragma unroll
    for (int t = 1; t < 8; ++t) m = fmaxf(m, acc[t][r]);
    rm[r] = m;
  }
#pragma unroll
  for (int off = 1; off < 16; off <<= 1)
#pragma unroll
    for (int r = 0; r < 8; ++r) rm[r] = fmaxf(rm[r], __shfl_xor(rm[r], off, 32));
  if (nl == 0)
#pragma unroll
    for (int r = 0; r < 8; ++r) red[wave][half * 8 + r] = rm[r];
  __syncthreads();
  float gmax[8];
#pragma unroll
  for (int r = 0; r < 8; ++r) {
    float m = -3.4e38f;
#pragma unroll
    for (int w2 = 0; w2 < 8; ++w2) m = fmaxf(m, red[w2][half * 8 + r]);
    gmax[r] = m;
  }
  __syncthreads();

  // ---- exp + row sum ----
#pragma unroll
  for (int t = 0; t < 8; ++t)
#pragma unroll
    for (int r = 0; r < 8; ++r) acc[t][r] = __expf(acc[t][r] - gmax[r]);
  float rs[8];
#pragma unroll
  for (int r = 0; r < 8; ++r) {
    float s = acc[0][r];
#pragma unroll
    for (int t = 1; t < 8; ++t) s += acc[t][r];
    rs[r] = s;
  }
#pragma unroll
  for (int off = 1; off < 16; off <<= 1)
#pragma unroll
    for (int r = 0; r < 8; ++r) rs[r] += __shfl_xor(rs[r], off, 32);
  if (nl == 0)
#pragma unroll
    for (int r = 0; r < 8; ++r) red[wave][half * 8 + r] = rs[r];
  __syncthreads();
  float inv[8];
#pragma unroll
  for (int r = 0; r < 8; ++r) {
    float s = 0.0f;
#pragma unroll
    for (int w2 = 0; w2 < 8; ++w2) s += red[w2][half * 8 + r];
    inv[r] = 1.0f / s;
  }

  // ---- normalize: write fp32 weights to d_out, stage bf16 copy in LDS ----
#pragma unroll
  for (int t = 0; t < 8; ++t) {
    const int ncol = wave * 128 + t * 16 + nl;
#pragma unroll
    for (int r = 0; r < 8; ++r) {
      const int mrow = half * 8 + r;
      const float wv = acc[t][r] * inv[r];
      Wout[(bcBase + q0 + mrow) * SS + ncol] = wv;
      wlds[wave][mrow][t * 16 + nl] = (__bf16)wv;
    }
  }
  __syncthreads();

  // ---- attended: (16 q) x (128 keys per wave) @ V strip -> partial 16x16 ----
  v8f av = {};
#pragma unroll
  for (int kt = 0; kt < 4; ++kt) {
    v16bf aw;
    const int kb = kt * 32 + half * 16;
    ((v8bf*)&aw)[0] = *(const v8bf*)&wlds[wave][nl][kb];
    ((v8bf*)&aw)[1] = *(const v8bf*)&wlds[wave][nl][kb + 8];
    const __bf16* vrow = Vt + (((size_t)b * CC + c) * HH + nl) * SS +
                         wave * 128 + kt * 32 + half * 16;
    v16bf bv = *(const v16bf*)vrow;
    av = wmma_bf16(aw, bv, av);
  }
#pragma unroll
  for (int r = 0; r < 8; ++r) attacc[wave][half * 8 + r][nl] = av[r];
  __syncthreads();

  // ---- cross-wave sum + write merged bf16 ----
  {
    const int M = tid >> 4, N = tid & 15;  // 256 threads = 16x16 tile
    float s = 0.0f;
#pragma unroll
    for (int w2 = 0; w2 < 8; ++w2) s += attacc[w2][M][N];
    const size_t row = (size_t)b * SS + q0 + M;
    Merged[row * DD + c * 16 + N] = (__bf16)s;
  }
}

// ---------------------------------------------------------------------------
// Host-side orchestration
// ---------------------------------------------------------------------------
extern "C" void kernel_launch(void* const* d_in, const int* in_sizes, int n_in,
                              void* d_out, int out_size, void* d_ws,
                              size_t ws_size, hipStream_t stream) {
  const float* Q  = (const float*)d_in[0];
  const float* K  = (const float*)d_in[1];
  const float* V  = (const float*)d_in[2];
  const float* Wq = (const float*)d_in[3];
  const float* Wk = (const float*)d_in[4];
  const float* Wv = (const float*)d_in[5];
  const float* Wo = (const float*)d_in[6];

  float* out     = (float*)d_out;
  float* weights = out + (size_t)NB * SS * DD;  // 2M floats in, then [B,C,S,S]

  char* ws = (char*)d_ws;
  const size_t MB = 1024 * 1024;
  __bf16* Qbf  = (__bf16*)(ws + 0 * MB);   // 4MB each for Q/K/V bf16
  __bf16* Kbf  = (__bf16*)(ws + 4 * MB);
  __bf16* Vbf  = (__bf16*)(ws + 8 * MB);
  __bf16* Wqb  = (__bf16*)(ws + 12 * MB);  // 2MB each weight
  __bf16* Wkb  = (__bf16*)(ws + 14 * MB);
  __bf16* Wvb  = (__bf16*)(ws + 16 * MB);
  __bf16* Wob  = (__bf16*)(ws + 18 * MB);
  __bf16* QpA  = (__bf16*)(ws + 20 * MB);  // 8MB  [B][C][S][HP]
  __bf16* KpA  = (__bf16*)(ws + 28 * MB);  // 8MB
  __bf16* VpT  = (__bf16*)(ws + 36 * MB);  // 4MB  [B][C][H][S]
  __bf16* Mrg  = (__bf16*)(ws + 40 * MB);  // 4MB  [B*S][D]

  const int nQ = NB * SS * DD;  // 2M
  const int nW = DD * DD;       // 1M
  cvt_f32_to_bf16<<<2048, 256, 0, stream>>>(Q, Qbf, nQ);
  cvt_f32_to_bf16<<<2048, 256, 0, stream>>>(K, Kbf, nQ);
  cvt_f32_to_bf16<<<2048, 256, 0, stream>>>(V, Vbf, nQ);
  cvt_f32_to_bf16<<<1024, 256, 0, stream>>>(Wq, Wqb, nW);
  cvt_f32_to_bf16<<<1024, 256, 0, stream>>>(Wk, Wkb, nW);
  cvt_f32_to_bf16<<<1024, 256, 0, stream>>>(Wv, Wvb, nW);
  cvt_f32_to_bf16<<<1024, 256, 0, stream>>>(Wo, Wob, nW);

  dim3 gGemm(MR / 64, DD / 128);  // 32 x 8
  gemm_xt<1><<<gGemm, 256, 0, stream>>>(Qbf, Wqb, nullptr, QpA);
  gemm_xt<1><<<gGemm, 256, 0, stream>>>(Kbf, Wkb, nullptr, KpA);
  gemm_xt<2><<<gGemm, 256, 0, stream>>>(Vbf, Wvb, nullptr, VpT);

  dim3 gAttn(SS / 16, CC, NB);  // 64 x 64 x 2
  attn_fused<<<gAttn, 256, 0, stream>>>(QpA, KpA, VpT, weights, Mrg);

  gemm_xt<0><<<gGemm, 256, 0, stream>>>(Mrg, Wob, out, nullptr);
}
